// MultiHeadAttention_21663815041695
// MI455X (gfx1250) — matrix operand order repaired and checked by reading, not compiled, not run on hardware
//
#include <hip/hip_runtime.h>

// ---------------------------------------------------------------------------
// MI455X (gfx1250) multi-head attention forward.
//   out = softmax( (XQ Wq^T + bq)(XK Wk^T + bk)^T / sqrt(dk) + mask ) (XV Wv^T + bv) Wo^T + bo
// All GEMMs run on v_wmma_f32_16x16x32_bf16 (wave32), inputs converted
// f32 -> bf16 while staging into LDS. Attention is a flash-style kernel.
// ---------------------------------------------------------------------------

typedef __attribute__((ext_vector_type(16))) __bf16 v16bf;
typedef __attribute__((ext_vector_type(8)))  __bf16 v8bf;
typedef __attribute__((ext_vector_type(8)))  float  v8f;
typedef __attribute__((ext_vector_type(4)))  float  v4f;

#define D_MODEL 1024
#define NHEAD   16
#define DKH     64
#define BATCH   2
#define SEQ     2048
#define MROWS   (BATCH * SEQ)   // 4096

// ---- CDNA5 WMMA 16x16x32 bf16 fragment loaders (ISA 7.12.2 layouts) -------
// A (16xK=32, 16-bit): lane m=l&15 holds its row; VGPR0-3 = K {0..7 | 8..15}
// by lane half, VGPR4-7 = +16.
__device__ inline v16bf load_frag_a(const __bf16* p, int ld, int lane) {
  const __bf16* r = p + (size_t)(lane & 15) * ld + ((lane >> 4) << 3);
  v8bf lo = *(const v8bf*)(r);        // k = khalf .. khalf+7
  v8bf hi = *(const v8bf*)(r + 16);   // k = 16+khalf .. 16+khalf+7
  v16bf out;
#pragma unroll
  for (int i = 0; i < 8; ++i) { out[i] = lo[i]; out[i + 8] = hi[i]; }
  return out;
}

// B (K=32 x 16, 16-bit): lane n=l&15 holds its column; VGPR0-3 = K {0..7 | 8..15}
// by lane half, VGPR4-7 = +16. Source array is [n][k] row-major.
__device__ inline v16bf load_frag_b(const __bf16* p, int ld, int lane) {
  const __bf16* r = p + (size_t)(lane & 15) * ld + ((lane >> 4) << 3);
  v8bf lo = *(const v8bf*)(r);        // k = khalf .. khalf+7
  v8bf hi = *(const v8bf*)(r + 16);   // k = 16+khalf .. 16+khalf+7
  v16bf out;
#pragma unroll
  for (int i = 0; i < 8; ++i) { out[i] = lo[i]; out[i + 8] = hi[i]; }
  return out;
}

// C/D f32: lane n=l&15 is the column; row m = 8*(l>>4) + vgpr.

// ---- 8-element load + convert-to-bf16 helpers ------------------------------
__device__ inline v8bf cvt8(const float* p) {
  v4f a = *(const v4f*)p;
  v4f b = *(const v4f*)(p + 4);
  v8bf r;
#pragma unroll
  for (int i = 0; i < 4; ++i) { r[i] = (__bf16)a[i]; r[i + 4] = (__bf16)b[i]; }
  return r;
}
__device__ inline v8bf cvt8(const __bf16* p) { return *(const v8bf*)p; }

// ---------------------------------------------------------------------------
// GEMM: C(M x N) = A(M x K) * W(N x K)^T + bias(N).  M=4096, N=K=1024.
// MODE 0: write f32 row-major [M][N]     (final output projection)
// MODE 1: write bf16 split-head [B][H][S][dk]  (q/k/v projections)
// Workgroup: 256 thr = 8 waves; tile 64(M) x 128(N); K step 32.
// ---------------------------------------------------------------------------
template <typename AT, int MODE>
__global__ void __launch_bounds__(256)
gemm_bias_kernel(const AT* __restrict__ A, const float* __restrict__ W,
                 const float* __restrict__ bias, void* __restrict__ Cout) {
  __shared__ __attribute__((aligned(16))) __bf16 ldsA[64][40];    // padded rows
  __shared__ __attribute__((aligned(16))) __bf16 ldsB[128][40];

  const int tid  = threadIdx.x;
  const int lane = tid & 31;
  const int wave = tid >> 5;
  const int mblk = blockIdx.y * 64;
  const int nblk = blockIdx.x * 128;
  const int mw   = (wave >> 1) * 16;   // wave's M offset in tile
  const int nw   = (wave & 1) * 64;    // wave's N offset in tile

  const int rowA = tid >> 2, colA = (tid & 3) * 8;     // 64 x 32 A tile
  const int rowB = tid >> 1, colB = (tid & 1) * 16;    // 128 x 32 W tile

  v8f acc[4];
#pragma unroll
  for (int t = 0; t < 4; ++t)
#pragma unroll
    for (int i = 0; i < 8; ++i) acc[t][i] = 0.f;

  for (int k0 = 0; k0 < D_MODEL; k0 += 32) {
    // stage (convert to bf16 on the fly)
    *(v8bf*)&ldsA[rowA][colA] =
        cvt8(A + (size_t)(mblk + rowA) * D_MODEL + k0 + colA);
    *(v8bf*)&ldsB[rowB][colB] =
        cvt8(W + (size_t)(nblk + rowB) * D_MODEL + k0 + colB);
    *(v8bf*)&ldsB[rowB][colB + 8] =
        cvt8(W + (size_t)(nblk + rowB) * D_MODEL + k0 + colB + 8);
    if (k0 + 32 < D_MODEL) {  // global_prefetch_b8 for the next K tile
      __builtin_prefetch(A + (size_t)(mblk + rowA) * D_MODEL + k0 + 32 + colA, 0, 3);
      __builtin_prefetch(W + (size_t)(nblk + rowB) * D_MODEL + k0 + 32 + colB, 0, 3);
    }
    __syncthreads();

    v16bf af = load_frag_a(&ldsA[mw][0], 40, lane);
#pragma unroll
    for (int t = 0; t < 4; ++t) {
      v16bf bf = load_frag_b(&ldsB[nw + t * 16][0], 40, lane);
      acc[t] = __builtin_amdgcn_wmma_f32_16x16x32_bf16(
          false, af, false, bf, (short)0, acc[t], false, false);
    }
    __syncthreads();
  }

  // epilogue
  const int cn  = lane & 15;
  const int cm0 = (lane >> 4) * 8;
#pragma unroll
  for (int t = 0; t < 4; ++t) {
    const int gn = nblk + nw + t * 16 + cn;
    const float bv = bias[gn];
#pragma unroll
    for (int v = 0; v < 8; ++v) {
      const int gm = mblk + mw + cm0 + v;
      const float val = acc[t][v] + bv;
      if (MODE == 0) {
        ((float*)Cout)[(size_t)gm * D_MODEL + gn] = val;
      } else {
        const int b = gm >> 11, s = gm & (SEQ - 1);
        const int h = gn >> 6,  d = gn & (DKH - 1);
        ((__bf16*)Cout)[(((size_t)(b * NHEAD + h) * SEQ) + s) * DKH + d] =
            (__bf16)val;
      }
    }
  }
}

// ---------------------------------------------------------------------------
// Flash attention: per (b,h), q-tile of 64 rows per workgroup (4 waves x 16),
// kv blocks of 64. scores = q k^T / 8, masked, online softmax, o += p v.
// ---------------------------------------------------------------------------
__global__ void __launch_bounds__(128)
flash_attn_kernel(const __bf16* __restrict__ qh, const __bf16* __restrict__ kh,
                  const __bf16* __restrict__ vh, const int* __restrict__ mask,
                  __bf16* __restrict__ out) {
  __shared__ __attribute__((aligned(16))) __bf16 Kt[64][72];        // [kv][dk]
  __shared__ __attribute__((aligned(16))) __bf16 Vt[64][72];        // [dk][kv]
  __shared__ __attribute__((aligned(16))) __bf16 Pld[4][16][72];    // per wave

  const int tid = threadIdx.x, lane = tid & 31, wave = tid >> 5;
  const int bh = blockIdx.y, b = bh >> 4, h = bh & (NHEAD - 1);
  const int qrow0 = blockIdx.x * 64 + wave * 16;

  const __bf16* qptr  = qh + ((size_t)bh * SEQ + qrow0) * DKH;
  const __bf16* kbase = kh + (size_t)bh * SEQ * DKH;
  const __bf16* vbase = vh + (size_t)bh * SEQ * DKH;

  const v16bf qf0 = load_frag_a(qptr, DKH, lane);        // dk 0..31
  const v16bf qf1 = load_frag_a(qptr + 32, DKH, lane);   // dk 32..63

  const int cn = lane & 15, cm0 = (lane >> 4) * 8;
  float mrow[8], lrow[8];
  v8f oacc[4];
#pragma unroll
  for (int v = 0; v < 8; ++v) { mrow[v] = -1e30f; lrow[v] = 0.f; }
#pragma unroll
  for (int u = 0; u < 4; ++u)
#pragma unroll
    for (int i = 0; i < 8; ++i) oacc[u][i] = 0.f;

  const int sr = tid >> 1, sc = (tid & 1) * 32;   // staging coords

  for (int kv0 = 0; kv0 < SEQ; kv0 += 64) {
    // --- stage K (row-major) and V (transposed) into LDS ---
    const __bf16* kg = kbase + (size_t)(kv0 + sr) * DKH + sc;
#pragma unroll
    for (int j = 0; j < 4; ++j)
      *(v8bf*)&Kt[sr][sc + j * 8] = *(const v8bf*)(kg + j * 8);
    const __bf16* vg = vbase + (size_t)(kv0 + sr) * DKH + sc;
#pragma unroll
    for (int j = 0; j < 32; ++j) Vt[sc + j][sr] = vg[j];
    __syncthreads();

    // --- scores: 16 x 64, dk contraction via 2 chained WMMAs ---
    v8f s[4];
#pragma unroll
    for (int t = 0; t < 4; ++t) {
#pragma unroll
      for (int i = 0; i < 8; ++i) s[t][i] = 0.f;
      v16bf bk0 = load_frag_b(&Kt[t * 16][0], 72, lane);
      s[t] = __builtin_amdgcn_wmma_f32_16x16x32_bf16(
          false, qf0, false, bk0, (short)0, s[t], false, false);
      v16bf bk1 = load_frag_b(&Kt[t * 16][32], 72, lane);
      s[t] = __builtin_amdgcn_wmma_f32_16x16x32_bf16(
          false, qf1, false, bk1, (short)0, s[t], false, false);
    }

    // --- scale, mask, online softmax (rows split per C layout) ---
#pragma unroll
    for (int v = 0; v < 8; ++v) {
      const int qr = qrow0 + cm0 + v;
      float mx = -1e30f;
#pragma unroll
      for (int t = 0; t < 4; ++t) {
        float sv = s[t][v] * 0.125f;   // 1/sqrt(64)
        const int kc = kv0 + t * 16 + cn;
        const int mv = mask[((size_t)b * SEQ + qr) * SEQ + kc];
        sv = (mv == 0) ? -1e9f : sv;
        s[t][v] = sv;
        mx = fmaxf(mx, sv);
      }
#pragma unroll
      for (int i = 1; i < 16; i <<= 1) mx = fmaxf(mx, __shfl_xor(mx, i, 32));
      const float mnew  = fmaxf(mrow[v], mx);
      const float alpha = __expf(mrow[v] - mnew);
      float rs = 0.f;
#pragma unroll
      for (int t = 0; t < 4; ++t) {
        const float p = __expf(s[t][v] - mnew);
        s[t][v] = p;
        rs += p;
      }
#pragma unroll
      for (int i = 1; i < 16; i <<= 1) rs += __shfl_xor(rs, i, 32);
      lrow[v] = lrow[v] * alpha + rs;
      mrow[v] = mnew;
#pragma unroll
      for (int u = 0; u < 4; ++u) oacc[u][v] *= alpha;
    }

    // --- round-trip P through per-wave LDS: C layout -> A layout ---
#pragma unroll
    for (int t = 0; t < 4; ++t)
#pragma unroll
      for (int v = 0; v < 8; ++v)
        Pld[wave][cm0 + v][t * 16 + cn] = (__bf16)s[t][v];
    // same-wave DS ops are in-order; compiler inserts s_wait_dscnt for reads

    const v16bf pf0 = load_frag_a(&Pld[wave][0][0], 72, lane);    // kv 0..31
    const v16bf pf1 = load_frag_a(&Pld[wave][0][32], 72, lane);   // kv 32..63
#pragma unroll
    for (int u = 0; u < 4; ++u) {
      v16bf vf0 = load_frag_b(&Vt[u * 16][0], 72, lane);
      oacc[u] = __builtin_amdgcn_wmma_f32_16x16x32_bf16(
          false, pf0, false, vf0, (short)0, oacc[u], false, false);
      v16bf vf1 = load_frag_b(&Vt[u * 16][32], 72, lane);
      oacc[u] = __builtin_amdgcn_wmma_f32_16x16x32_bf16(
          false, pf1, false, vf1, (short)0, oacc[u], false, false);
    }
    __syncthreads();
  }

  // --- normalize and write out, layout [B][S][H*dk] (ready for Wo GEMM) ---
#pragma unroll
  for (int u = 0; u < 4; ++u)
#pragma unroll
    for (int v = 0; v < 8; ++v) {
      const int qr = qrow0 + cm0 + v;
      const int d  = u * 16 + cn;
      const float o = oacc[u][v] / lrow[v];
      out[((size_t)b * SEQ + qr) * D_MODEL + h * DKH + d] = (__bf16)o;
    }
}

// ---------------------------------------------------------------------------
extern "C" void kernel_launch(void* const* d_in, const int* in_sizes, int n_in,
                              void* d_out, int out_size, void* d_ws,
                              size_t ws_size, hipStream_t stream) {
  (void)in_sizes; (void)n_in; (void)out_size; (void)ws_size;
  const float* Q    = (const float*)d_in[0];
  const float* K    = (const float*)d_in[1];
  const float* V    = (const float*)d_in[2];
  const int*   mask = (const int*)  d_in[3];
  const float* Wq   = (const float*)d_in[4];
  const float* bq   = (const float*)d_in[5];
  const float* Wk   = (const float*)d_in[6];
  const float* bk   = (const float*)d_in[7];
  const float* Wv   = (const float*)d_in[8];
  const float* bv   = (const float*)d_in[9];
  const float* Wo   = (const float*)d_in[10];
  const float* bo   = (const float*)d_in[11];

  const size_t ESZ = (size_t)MROWS * D_MODEL;   // 4M elements
  __bf16* qhp  = (__bf16*)d_ws;                 // [B][H][S][dk] bf16
  __bf16* khp  = qhp + ESZ;
  __bf16* vhp  = khp + ESZ;
  __bf16* attn = vhp + ESZ;                     // [B][S][H*dk] bf16

  const dim3 gg(D_MODEL / 128, MROWS / 64);     // (8, 64), 256 thr
  gemm_bias_kernel<float, 1><<<gg, 256, 0, stream>>>(Q, Wq, bq, qhp);
  gemm_bias_kernel<float, 1><<<gg, 256, 0, stream>>>(K, Wk, bk, khp);
  gemm_bias_kernel<float, 1><<<gg, 256, 0, stream>>>(V, Wv, bv, vhp);

  flash_attn_kernel<<<dim3(SEQ / 64, BATCH * NHEAD), 128, 0, stream>>>(
      qhp, khp, vhp, mask, attn);

  gemm_bias_kernel<__bf16, 0><<<gg, 256, 0, stream>>>(attn, Wo, bo,
                                                      (float*)d_out);
}
